// Attention_20444044329649
// MI455X (gfx1250) — matrix operand order repaired and checked by reading, not compiled, not run on hardware
//
#include <hip/hip_runtime.h>
#include <hip/hip_bf16.h>
#include <stdint.h>

// ---------------------------------------------------------------------------
// Shapes
// ---------------------------------------------------------------------------
#define Bn   32
#define Hn   32
#define Wn   32
#define Cn   512
#define Nn   (Hn * Wn)          // 1024 tokens per batch
#define Mtot (Bn * Nn)          // 32768 rows total
#define C3   (3 * Cn)           // 1536
#define NGRP 32
#define GSZ  (Cn / NGRP)        // 16 channels per group
#define EPSV 1e-5f

typedef __attribute__((ext_vector_type(16))) __bf16 v16bf;
typedef __attribute__((ext_vector_type(8)))  __bf16 v8bf;
typedef __attribute__((ext_vector_type(8)))  float  v8f;
typedef __attribute__((ext_vector_type(4)))  unsigned int u32x4;
typedef __attribute__((ext_vector_type(8)))  int i32x8;
typedef __attribute__((ext_vector_type(4)))  int i32x4;

__device__ __forceinline__ v8f v8f_zero() {
    v8f v;
#pragma unroll
    for (int i = 0; i < 8; ++i) v[i] = 0.0f;
    return v;
}

__device__ __forceinline__ v8f wmma_bf16(v16bf a, v16bf b, v8f c) {
    // D = A(16x32 bf16) x B(32x16 bf16) + C(16x16 f32)
    return __builtin_amdgcn_wmma_f32_16x16x32_bf16(
        false, a, false, b, (short)0, c, false, false);
}

// A fragment (16x32, 16-bit): lane m=lane&15; half=lane>>4;
// element e -> K = e + 16*half (one contiguous 32B run)
__device__ __forceinline__ v16bf lds_load_a(const __bf16* p, int stride,
                                            int row0, int k0) {
    const int lane = threadIdx.x & 31;
    const int half = lane >> 4;
    const int m    = lane & 15;
    const __bf16* base = p + (row0 + m) * stride + k0 + half * 16;
    return *(const v16bf*)(base);
}

// B fragment (32x16, 16-bit) from a K-TRANSPOSED LDS image p[n][k]:
// lane n=lane&15, element e -> K = e + 16*half: one contiguous 32B run.
__device__ __forceinline__ v16bf lds_load_bt(const __bf16* p, int stride,
                                             int n0, int k0) {
    const int lane = threadIdx.x & 31;
    const int half = lane >> 4;
    const int n    = lane & 15;
    return *(const v16bf*)(p + (n0 + n) * stride + k0 + half * 16);
}

// ---------------------------------------------------------------------------
// Tensor Data Mover: 2D bf16 tile (d0 contiguous elems x d1 rows, row stride
// stride0 elems) global -> LDS byte offset lds_off. EXEC ignored; TENSORcnt.
// ---------------------------------------------------------------------------
__device__ __forceinline__ void tdm_load_2d(unsigned lds_off, const void* gptr,
                                            unsigned d0, unsigned d1,
                                            unsigned stride0) {
    const unsigned long long ga = (unsigned long long)(uintptr_t)gptr;
    u32x4 g0;
    g0[0] = 1u;                                        // count = 1 (valid)
    g0[1] = lds_off;                                   // lds_addr
    g0[2] = (unsigned)ga;                              // global_addr[31:0]
    g0[3] = (unsigned)((ga >> 32) & 0x1FFFFFFu)        // global_addr[56:32]
            | (2u << 30);                              // type = 2 (image)
    i32x8 g1;
    g1[0] = (int)(1u << 16);                           // data_size = 2 bytes
    g1[1] = (int)((d0 & 0xFFFFu) << 16);               // tensor_dim0 lo16
    g1[2] = (int)(((d0 >> 16) & 0xFFFFu) | ((d1 & 0xFFFFu) << 16));
    g1[3] = (int)(((d1 >> 16) & 0xFFFFu) | ((d0 & 0xFFFFu) << 16)); // tile_dim0
    g1[4] = (int)(d1 & 0xFFFFu);                       // tile_dim1 (dim2=0)
    g1[5] = (int)stride0;                              // dim0_stride lo32
    g1[6] = 0;
    g1[7] = 0;
    i32x4 z4 = {0, 0, 0, 0};
#if __has_include(<hip/amd_detail/amd_gfx1250_TDM.h>)
    i32x8 z8 = {0, 0, 0, 0, 0, 0, 0, 0};
    __builtin_amdgcn_tensor_load_to_lds(g0, g1, z4, z4, z8, 0);
#else
    __builtin_amdgcn_tensor_load_to_lds(g0, g1, z4, z4, 0);
#endif
}

// ---------------------------------------------------------------------------
// Kernel 1: per-(batch, group) mean / rstd
// ---------------------------------------------------------------------------
__global__ void gn_stats_kernel(const float* __restrict__ x,
                                float* __restrict__ stats) {
    const int bg = blockIdx.x;            // 0..1023
    const int b  = bg >> 5;
    const int g  = bg & 31;
    const float* xb = x + (size_t)b * Nn * Cn + g * GSZ;

    float s = 0.0f, s2 = 0.0f;
    for (int i = threadIdx.x; i < Nn * GSZ; i += 256) {
        const int sp = i >> 4;
        const int cc = i & 15;
        const float v = xb[(size_t)sp * Cn + cc];
        s += v;
        s2 += v * v;
    }
    __shared__ float red0[256];
    __shared__ float red1[256];
    red0[threadIdx.x] = s;
    red1[threadIdx.x] = s2;
    __syncthreads();
    for (int off = 128; off > 0; off >>= 1) {
        if ((int)threadIdx.x < off) {
            red0[threadIdx.x] += red0[threadIdx.x + off];
            red1[threadIdx.x] += red1[threadIdx.x + off];
        }
        __syncthreads();
    }
    if (threadIdx.x == 0) {
        const float inv = 1.0f / (float)(Nn * GSZ);
        const float mean = red0[0] * inv;
        const float var  = red1[0] * inv - mean * mean;
        stats[bg * 2 + 0] = mean;
        stats[bg * 2 + 1] = rsqrtf(var + EPSV);
    }
}

// ---------------------------------------------------------------------------
// Kernel 2: fused GroupNorm + QKV GEMM (M=32768, K=512, N=1536) -> bf16
// ---------------------------------------------------------------------------
__global__ void gn_qkv_kernel(const float* __restrict__ x,
                              const float* __restrict__ stats,
                              const float* __restrict__ gn_scale,
                              const float* __restrict__ gn_bias,
                              const float* __restrict__ w_qkv,
                              const float* __restrict__ b_qkv,
                              __bf16* __restrict__ qkv) {
    __shared__ __bf16 As[128 * 32];   // normalized activations [row][k]
    __shared__ __bf16 Bst[64 * 32];   // weight tile TRANSPOSED [n][k]

    const int n0 = blockIdx.x * 64;
    const int m0 = blockIdx.y * 128;
    const int tid  = threadIdx.x;
    const int wave = tid >> 5;
    const int lane = tid & 31;
    const int half = lane >> 4;
    const int nl   = lane & 15;

    v8f acc[4];
#pragma unroll
    for (int t = 0; t < 4; ++t) acc[t] = v8f_zero();

    for (int kk = 0; kk < Cn / 32; ++kk) {
        __syncthreads();
        for (int i = tid; i < 128 * 32; i += 256) {
            const int row = i >> 5;
            const int col = i & 31;
            const int m = m0 + row;
            const int c = kk * 32 + col;
            const int b = m >> 10;
            const int g = c >> 4;
            const float mean = stats[(b * NGRP + g) * 2 + 0];
            const float rstd = stats[(b * NGRP + g) * 2 + 1];
            const float v = (x[(size_t)m * Cn + c] - mean) * rstd *
                            gn_scale[c] + gn_bias[c];
            As[row * 32 + col] = (__bf16)v;
        }
        for (int i = tid; i < 32 * 64; i += 256) {
            const int row = i >> 6;   // k
            const int col = i & 63;   // n
            Bst[col * 32 + row] =
                (__bf16)w_qkv[(size_t)(kk * 32 + row) * C3 + n0 + col];
        }
        __syncthreads();

        // batch all fragment loads, then issue 4 independent WMMAs
        const v16bf a = lds_load_a(As, 32, wave * 16, 0);
        v16bf bfr[4];
#pragma unroll
        for (int t = 0; t < 4; ++t) bfr[t] = lds_load_bt(Bst, 32, t * 16, 0);
#pragma unroll
        for (int t = 0; t < 4; ++t) acc[t] = wmma_bf16(a, bfr[t], acc[t]);
    }

#pragma unroll
    for (int t = 0; t < 4; ++t) {
#pragma unroll
        for (int r = 0; r < 8; ++r) {
            const int row = m0 + wave * 16 + r + half * 8;
            const int col = n0 + t * 16 + nl;
            qkv[(size_t)row * C3 + col] = (__bf16)(acc[t][r] + b_qkv[col]);
        }
    }
}

// ---------------------------------------------------------------------------
// Kernel 3: flash attention, one (batch, 128-query-row tile) per block.
// Dynamic LDS carve-up (TDM descriptors use these byte offsets):
//   Qs @ 0       : 128x512 bf16 (TDM)
//   K0 @ 131072  :  32x512 bf16 (TDM, ping)
//   K1 @ 163840  :  32x512 bf16 (TDM, pong)
//   Vt @ 196608  : 512x32  bf16 [c][key] (manual transposed staging)
//   Ps @ 229376  : 8 waves x 16x32 bf16
// ---------------------------------------------------------------------------
#define QS_OFF 0u
#define K0_OFF (128u * Cn * 2u)
#define K1_OFF (K0_OFF + 32u * Cn * 2u)
#define VT_OFF (K1_OFF + 32u * Cn * 2u)
#define PS_OFF (VT_OFF + 32u * Cn * 2u)
#define ATTN_LDS (PS_OFF + 8u * 16u * 32u * 2u)   // 232 KB

__global__ void attn_kernel(const __bf16* __restrict__ qkv,
                            __bf16* __restrict__ aout) {
    extern __shared__ char smem[];
    __bf16* Qs = (__bf16*)(smem + QS_OFF);
    __bf16* Vt = (__bf16*)(smem + VT_OFF);
    __bf16* Ps = (__bf16*)(smem + PS_OFF);

    const int b  = blockIdx.x >> 3;
    const int qt = blockIdx.x & 7;
    const __bf16* Q = qkv + ((size_t)b * Nn + qt * 128) * C3;
    const __bf16* K = qkv + (size_t)b * Nn * C3 + Cn;
    const __bf16* V = qkv + (size_t)b * Nn * C3 + 2 * Cn;

    const int tid  = threadIdx.x;
    const int wave = tid >> 5;
    const int lane = tid & 31;
    const int half = lane >> 4;
    const int nl   = lane & 15;

    // prologue DMAs: whole Q tile + first K block (both in flight)
    if (wave == 0) {
        tdm_load_2d(QS_OFF, Q, Cn, 128, C3);
        tdm_load_2d(K0_OFF, K, Cn, 32, C3);
    }

    v8f o[32];
#pragma unroll
    for (int t = 0; t < 32; ++t) o[t] = v8f_zero();
    float rm[8], rl[8];
#pragma unroll
    for (int r = 0; r < 8; ++r) { rm[r] = -1e30f; rl[r] = 0.0f; }

    const float sc = 0.04419417382415922f;    // 1/sqrt(512)
    __bf16* pw = Ps + wave * (16 * 32);

    for (int j = 0; j < Nn / 32; ++j) {       // 32 key blocks of 32
        __syncthreads();                      // prev compute done (Vt, Kbuf)
        // software pipeline: start the NEXT K block DMA before waiting
        if (wave == 0 && j + 1 < Nn / 32)
            tdm_load_2d(((j + 1) & 1) ? K1_OFF : K0_OFF,
                        K + (size_t)(j + 1) * 32 * C3, Cn, 32, C3);
        // V block: staged transposed Vt[c][key] (overlaps the TDM DMAs)
        for (int i = tid; i < 32 * Cn / 8; i += 256) {
            const int key = i >> 6;
            const int c0  = (i & 63) * 8;
            uint4 d = *((const uint4*)(V + (size_t)(j * 32 + key) * C3) +
                        (i & 63));
            __bf16 tmp[8];
            *(uint4*)tmp = d;
#pragma unroll
            for (int e = 0; e < 8; ++e) Vt[(c0 + e) * 32 + key] = tmp[e];
        }
        if (j + 1 < Nn / 32)
            __builtin_prefetch(V + (size_t)(j + 1) * 32 * C3, 0, 1);
        // allow only the j+1 DMA to remain outstanding -> block j complete
        if (wave == 0) {
            if (j + 1 < Nn / 32) __builtin_amdgcn_s_wait_tensorcnt(1);
            else                 __builtin_amdgcn_s_wait_tensorcnt(0);
        }
        __syncthreads();

        // ping/pong K buffer (pointer computed per-use, NOT a pointer array)
        const __bf16* Ks = (const __bf16*)(smem + ((j & 1) ? K1_OFF : K0_OFF));
        const __bf16* QsW = Qs + wave * 16 * Cn;

        // S tile = Q(16x512) * K^T(512x32): batch loads (4 kc steps = 12
        // fragments) then 8 WMMAs on two independent accumulator chains
        v8f s0 = v8f_zero(), s1 = v8f_zero();
        for (int kc4 = 0; kc4 < Cn / 32; kc4 += 4) {
            v16bf a[4], b0[4], b1[4];
#pragma unroll
            for (int i = 0; i < 4; ++i) {
                const int k0 = (kc4 + i) * 32;
                a[i]  = lds_load_a(QsW, Cn, 0, k0);
                b0[i] = lds_load_bt(Ks, Cn, 0,  k0);
                b1[i] = lds_load_bt(Ks, Cn, 16, k0);
            }
#pragma unroll
            for (int i = 0; i < 4; ++i) {
                s0 = wmma_bf16(a[i], b0[i], s0);
                s1 = wmma_bf16(a[i], b1[i], s1);
            }
        }

        // online softmax (row M = r + 8*half lives across a 16-lane half)
        float al[8];
#pragma unroll
        for (int r = 0; r < 8; ++r) {
            s0[r] *= sc;
            s1[r] *= sc;
            float v = fmaxf(s0[r], s1[r]);
#pragma unroll
            for (int off = 1; off < 16; off <<= 1)
                v = fmaxf(v, __shfl_xor(v, off, 32));
            const float nm = fmaxf(rm[r], v);
            al[r] = __expf(rm[r] - nm);
            rm[r] = nm;
        }
#pragma unroll
        for (int r = 0; r < 8; ++r) {
            s0[r] = __expf(s0[r] - rm[r]);
            s1[r] = __expf(s1[r] - rm[r]);
            float v = s0[r] + s1[r];
#pragma unroll
            for (int off = 1; off < 16; off <<= 1)
                v += __shfl_xor(v, off, 32);
            rl[r] = rl[r] * al[r] + v;
        }
#pragma unroll
        for (int t = 0; t < 32; ++t)
#pragma unroll
            for (int r = 0; r < 8; ++r) o[t][r] *= al[r];

        // relayout P (C-layout -> A-fragment layout) through per-wave LDS
#pragma unroll
        for (int r = 0; r < 8; ++r) {
            pw[(r + half * 8) * 32 + nl]      = (__bf16)s0[r];
            pw[(r + half * 8) * 32 + 16 + nl] = (__bf16)s1[r];
        }
        asm volatile("s_wait_dscnt 0" ::: "memory");
        const v16bf pa = lds_load_a(pw, 32, 0, 0);

        // O(16x512) += P(16x32) * V(32x512): batches of 8 fragment loads
        // followed by 8 independent WMMA chains
#pragma unroll
        for (int t8 = 0; t8 < 32; t8 += 8) {
            v16bf bv[8];
#pragma unroll
            for (int i = 0; i < 8; ++i)
                bv[i] = lds_load_bt(Vt, 32, (t8 + i) * 16, 0);
#pragma unroll
            for (int i = 0; i < 8; ++i)
                o[t8 + i] = wmma_bf16(pa, bv[i], o[t8 + i]);
        }
    }

    float inv[8];
#pragma unroll
    for (int r = 0; r < 8; ++r) inv[r] = 1.0f / rl[r];
    __bf16* dst = aout + ((size_t)b * Nn + qt * 128 + wave * 16) * Cn;
#pragma unroll
    for (int t = 0; t < 32; ++t)
#pragma unroll
        for (int r = 0; r < 8; ++r)
            dst[(size_t)(r + half * 8) * Cn + t * 16 + nl] =
                (__bf16)(o[t][r] * inv[r]);
}

// ---------------------------------------------------------------------------
// Kernel 4: out = scores @ w_out + b_out + x  (M=32768, N=512, K=512)
// A tiles DMA'd by TDM, double-buffered; weight tile staged transposed.
// Dynamic LDS: As0 @ 0, As1 @ 8192 (128x32 each), Bst @ 16384 (64x32).
// ---------------------------------------------------------------------------
__global__ void outproj_kernel(const __bf16* __restrict__ aout,
                               const float* __restrict__ w_out,
                               const float* __restrict__ b_out,
                               const float* __restrict__ x,
                               float* __restrict__ out) {
    extern __shared__ char smem[];
    __bf16* Bst = (__bf16*)(smem + 16384);   // [n][k] 64x32

    const int n0 = blockIdx.x * 64;
    const int m0 = blockIdx.y * 128;
    const int tid  = threadIdx.x;
    const int wave = tid >> 5;
    const int lane = tid & 31;
    const int half = lane >> 4;
    const int nl   = lane & 15;

    v8f acc[4];
#pragma unroll
    for (int t = 0; t < 4; ++t) acc[t] = v8f_zero();

    if (wave == 0)
        tdm_load_2d(0u, aout + (size_t)m0 * Cn, 32, 128, Cn);

    for (int kk = 0; kk < Cn / 32; ++kk) {
        __syncthreads();
        if (wave == 0 && kk + 1 < Cn / 32)
            tdm_load_2d(((kk + 1) & 1) ? 8192u : 0u,
                        aout + (size_t)m0 * Cn + (kk + 1) * 32, 32, 128, Cn);
        for (int i = tid; i < 32 * 64; i += 256) {
            const int row = i >> 6;   // k
            const int col = i & 63;   // n
            Bst[col * 32 + row] =
                (__bf16)w_out[(size_t)(kk * 32 + row) * Cn + n0 + col];
        }
        if (wave == 0) {
            if (kk + 1 < Cn / 32) __builtin_amdgcn_s_wait_tensorcnt(1);
            else                  __builtin_amdgcn_s_wait_tensorcnt(0);
        }
        __syncthreads();

        // ping/pong A buffer (pointer computed per-use, NOT a pointer array)
        const __bf16* Acur = (const __bf16*)(smem + ((kk & 1) ? 8192u : 0u));
        const v16bf a = lds_load_a(Acur, 32, wave * 16, 0);
        v16bf bfr[4];
#pragma unroll
        for (int t = 0; t < 4; ++t) bfr[t] = lds_load_bt(Bst, 32, t * 16, 0);
#pragma unroll
        for (int t = 0; t < 4; ++t) acc[t] = wmma_bf16(a, bfr[t], acc[t]);
    }

#pragma unroll
    for (int t = 0; t < 4; ++t) {
#pragma unroll
        for (int r = 0; r < 8; ++r) {
            const int row = m0 + wave * 16 + r + half * 8;
            const int col = n0 + t * 16 + nl;
            out[(size_t)row * Cn + col] =
                acc[t][r] + b_out[col] + x[(size_t)row * Cn + col];
        }
    }
}

// ---------------------------------------------------------------------------
// Launch
// ---------------------------------------------------------------------------
extern "C" void kernel_launch(void* const* d_in, const int* in_sizes, int n_in,
                              void* d_out, int out_size, void* d_ws,
                              size_t ws_size, hipStream_t stream) {
    (void)in_sizes; (void)n_in; (void)out_size; (void)ws_size;
    const float* x        = (const float*)d_in[0];
    // d_in[1] = t (unused by the reference computation)
    const float* gn_scale = (const float*)d_in[2];
    const float* gn_bias  = (const float*)d_in[3];
    const float* w_qkv    = (const float*)d_in[4];
    const float* b_qkv    = (const float*)d_in[5];
    const float* w_out    = (const float*)d_in[6];
    const float* b_out    = (const float*)d_in[7];
    float* out = (float*)d_out;

    float*  stats = (float*)d_ws;                                   // 8 KB
    __bf16* qkv   = (__bf16*)((char*)d_ws + 8192);                  // 96 MB
    __bf16* aout  = (__bf16*)((char*)d_ws + 8192 +
                              (size_t)Mtot * C3 * sizeof(__bf16));  // 32 MB

    gn_stats_kernel<<<Bn * NGRP, 256, 0, stream>>>(x, stats);

    gn_qkv_kernel<<<dim3(C3 / 64, Mtot / 128), 256, 0, stream>>>(
        x, stats, gn_scale, gn_bias, w_qkv, b_qkv, qkv);

    attn_kernel<<<Bn * (Nn / 128), 256, ATTN_LDS, stream>>>(qkv, aout);

    outproj_kernel<<<dim3(Cn / 64, Mtot / 128), 256, 20480, stream>>>(
        aout, w_out, b_out, x, out);
}
